// SUNet_Level1_Block_11991548690663
// MI455X (gfx1250) — matrix operand order repaired and checked by reading, not compiled, not run on hardware
//
#include <hip/hip_runtime.h>

// ---------------------------------------------------------------------------
// CDNA5 WMMA types
// ---------------------------------------------------------------------------
typedef __attribute__((ext_vector_type(16))) __bf16 v16bf;
typedef __attribute__((ext_vector_type(8)))  float  v8f;

union FragBF { unsigned int u[8]; v16bf v; };

__device__ __forceinline__ unsigned short f2bf(float f) {
  unsigned int u = __float_as_uint(f);
  unsigned int r = u + 0x7FFFu + ((u >> 16) & 1u);   // round-to-nearest-even
  return (unsigned short)(r >> 16);
}

__device__ __forceinline__ float sigf(float x) { return 1.f / (1.f + __expf(-x)); }

// ---------------------------------------------------------------------------
// Global -> LDS 16-byte staging: async-to-LDS if the toolchain exposes it,
// otherwise plain b128 load + ds_store_b128.
// The builtin's pointer params are vector-of-4-int pointers in addrspace(1)
// (global) and addrspace(3) (LDS) -- per hipcc's diagnostic.
// ---------------------------------------------------------------------------
#if defined(__has_builtin)
#  if __has_builtin(__builtin_amdgcn_global_load_async_to_lds_b128) && \
      __has_builtin(__builtin_amdgcn_s_wait_asynccnt)
#    define ASYNC_LDS 1
#  endif
#endif
#ifndef ASYNC_LDS
#  define ASYNC_LDS 0
#endif

#if ASYNC_LDS
typedef int vi4_ld __attribute__((ext_vector_type(4)));
typedef __attribute__((address_space(1))) vi4_ld* gvec_t;
typedef __attribute__((address_space(3))) vi4_ld* lvec_t;
#endif

__device__ __forceinline__ void cp_b128(const void* g, void* l) {
#if ASYNC_LDS
  __builtin_amdgcn_global_load_async_to_lds_b128((gvec_t)g, (lvec_t)l, 0, 0);
#else
  *(uint4*)l = *(const uint4*)g;
#endif
}
__device__ __forceinline__ void cp_wait() {
#if ASYNC_LDS
  __builtin_amdgcn_s_wait_asynccnt(0);
#endif
}

// ---------------------------------------------------------------------------
// Weight / activation packing (fp32 -> bf16, conv-friendly layouts), done once
// per tensor instead of per block.
// ---------------------------------------------------------------------------
// [O=32][I=32][3][3] fp32 -> [tap][o][ci] bf16
__global__ void k_pack_w3(const float* __restrict__ w, unsigned short* __restrict__ o) {
  int i = blockIdx.x * blockDim.x + threadIdx.x;
  if (i >= 9 * 32 * 32) return;
  int ci = i % 32, oc = (i / 32) % 32, tap = i / (32 * 32);
  o[i] = f2bf(w[(oc * 32 + ci) * 9 + tap]);
}
// [32][128] fp32 -> bf16 (same layout)
__global__ void k_pack_w1(const float* __restrict__ w, unsigned short* __restrict__ o) {
  int i = blockIdx.x * blockDim.x + threadIdx.x;
  if (i < 32 * 128) o[i] = f2bf(w[i]);
}
// [N,C,H,W] fp32 -> [N,H,W,C] bf16
__global__ void k_pack_nhwc(const float* __restrict__ in, unsigned short* __restrict__ out,
                            int Cn, int HW, long long total) {
  long long i = (long long)blockIdx.x * blockDim.x + threadIdx.x;
  if (i >= total) return;
  int c = (int)(i % Cn);
  int hw = (int)((i / Cn) % HW);
  long long n = i / ((long long)Cn * HW);
  out[i] = f2bf(in[(n * Cn + c) * (long long)HW + hw]);
}

// ---------------------------------------------------------------------------
// 3x3 conv, Cin=Cout=32, pad=1, implicit GEMM on v_wmma_f32_16x16x32_bf16.
// Input is NHWC bf16 (so the LDS strip [row][x][c] is contiguous row segments
// staged with b128 / async-to-LDS copies). Output NCHW fp32.
// Block = 128 threads (4 waves); 32 outC x 32 pixels per block; 9 WMMA/wave.
// ---------------------------------------------------------------------------
__global__ void __launch_bounds__(128)
k_conv3x3_wmma(const unsigned short* __restrict__ xq, const unsigned short* __restrict__ wq,
               float* __restrict__ out, int H, int W) {
  __shared__ __align__(16) unsigned short w_s[9 * 32 * 32];   // [tap][o][ci]
  __shared__ __align__(16) unsigned short x_s[3 * 34 * 32];   // [row][xx][ci]
  const int tid = threadIdx.x;
  const int w0 = blockIdx.x * 32;
  const int h  = blockIdx.y;
  const int n  = blockIdx.z;

  for (int i = tid; i < 1152; i += 128)                       // 18 KB weights
    cp_b128((const uint4*)wq + i, (uint4*)w_s + i);

  const int xlo  = (w0 == 0) ? 0 : (w0 - 1);
  const int xhi  = min(W, w0 + 33);
  const int npix = xhi - xlo;
  const int lp0  = xlo - (w0 - 1);            // 1 iff left edge
  for (int r = 0; r < 3; ++r) {
    const int gh = h + r - 1;
    uint4* lrow = (uint4*)(x_s + r * 34 * 32);
    if (gh < 0 || gh >= H) {
      for (int i = tid; i < 34 * 4; i += 128) lrow[i] = make_uint4(0, 0, 0, 0);
    } else {
      const uint4* grow = (const uint4*)(xq + (((long long)n * H + gh) * W + xlo) * 32);
      uint4* ldst = (uint4*)(x_s + (r * 34 + lp0) * 32);
      for (int i = tid; i < npix * 4; i += 128) cp_b128(grow + i, ldst + i);
      if (lp0 && tid < 4) lrow[tid] = make_uint4(0, 0, 0, 0);                // xx = 0
      if (xhi < w0 + 33 && tid < 4) lrow[33 * 4 + tid] = make_uint4(0, 0, 0, 0); // xx = 33
    }
  }
  if (tid == 0 && h + 1 < H)
    __builtin_prefetch(xq + (((long long)n * H + h + 1) * W + w0) * 32, 0, 1);
  cp_wait();
  __syncthreads();

  const int lane = tid & 31, wave = tid >> 5;
  const int mh = wave & 1, nt = wave >> 1;
  const int half = lane >> 4, l16 = lane & 15;
  const int m = mh * 16 + l16;
  const int p = nt * 16 + l16;
  v8f acc = {};
  for (int tap = 0; tap < 9; ++tap) {
    const int kh = tap / 3, kw = tap % 3;
    FragBF fa, fb;
#pragma unroll
    for (int v = 0; v < 8; ++v) {
      const int kk = half * 16 + v * 2;       // A: K index
      fa.u[v] = *reinterpret_cast<const unsigned int*>(&w_s[(tap * 32 + m) * 32 + kk]);
      const int k2 = half * 16 + v * 2;                            // B: K index
      fb.u[v] = *reinterpret_cast<const unsigned int*>(&x_s[(kh * 34 + p + kw) * 32 + k2]);
    }
    acc = __builtin_amdgcn_wmma_f32_16x16x32_bf16(false, fa.v, false, fb.v,
                                                  (short)0, acc, false, false);
  }
  const long long imgBase = (long long)n * 32 * H * W;
  const int gw = w0 + p;
#pragma unroll
  for (int v = 0; v < 8; ++v) {
    const int mo = mh * 16 + v + 8 * half;
    out[imgBase + ((long long)mo * H + h) * W + gw] = acc[v];
  }
}

// ---------------------------------------------------------------------------
// 1x1 conv / pointwise GEMM: Cin=128, Cout=32. NHWC bf16 input; a 32-pixel
// strip is one contiguous 8 KB segment -> pure b128 / async staging.
// ---------------------------------------------------------------------------
__global__ void __launch_bounds__(128)
k_conv1x1_wmma(const unsigned short* __restrict__ xq, const unsigned short* __restrict__ wq,
               float* __restrict__ out, int H, int W, int relu) {
  __shared__ __align__(16) unsigned short w_s[32 * 128];  // [o][k]
  __shared__ __align__(16) unsigned short x_s[32 * 128];  // [pix][k]
  const int tid = threadIdx.x;
  const int w0 = blockIdx.x * 32;
  const int h  = blockIdx.y;
  const int n  = blockIdx.z;
  for (int i = tid; i < 512; i += 128)
    cp_b128((const uint4*)wq + i, (uint4*)w_s + i);
  const uint4* grow = (const uint4*)(xq + (((long long)n * H + h) * W + w0) * 128);
  for (int i = tid; i < 512; i += 128)
    cp_b128(grow + i, (uint4*)x_s + i);
  cp_wait();
  __syncthreads();

  const int lane = tid & 31, wave = tid >> 5;
  const int mh = wave & 1, nt = wave >> 1;
  const int half = lane >> 4, l16 = lane & 15;
  const int m = mh * 16 + l16;
  const int p = nt * 16 + l16;
  v8f acc = {};
  for (int step = 0; step < 4; ++step) {
    FragBF fa, fb;
#pragma unroll
    for (int v = 0; v < 8; ++v) {
      const int kk = step * 32 + half * 16 + v * 2;
      fa.u[v] = *reinterpret_cast<const unsigned int*>(&w_s[m * 128 + kk]);
      const int k2 = step * 32 + half * 16 + v * 2;
      fb.u[v] = *reinterpret_cast<const unsigned int*>(&x_s[p * 128 + k2]);
    }
    acc = __builtin_amdgcn_wmma_f32_16x16x32_bf16(false, fa.v, false, fb.v,
                                                  (short)0, acc, false, false);
  }
  const long long outBase = (long long)n * 32 * H * W;
  const int gw = w0 + p;
#pragma unroll
  for (int v = 0; v < 8; ++v) {
    const int mo = mh * 16 + v + 8 * half;
    float r = acc[v];
    if (relu) r = fmaxf(r, 0.f);
    out[outBase + ((long long)mo * H + h) * W + gw] = r;
  }
}

// ---------------------------------------------------------------------------
// Elementwise / recurrence kernels
// ---------------------------------------------------------------------------
__global__ void k_zero(float* p, int n) {
  int i = blockIdx.x * blockDim.x + threadIdx.x;
  if (i < n) p[i] = 0.f;
}

// multistep LIF: tau=2 (decay 0.5), hard reset; sequential in T per site.
__global__ void k_lif(const float* __restrict__ x, float* __restrict__ s,
                      long long n, int T, float vth) {
  long long i = (long long)blockIdx.x * blockDim.x + threadIdx.x;
  if (i >= n) return;
  float v = 0.f;
  for (int t = 0; t < T; ++t) {
    float h = v * 0.5f + x[(long long)t * n + i];
    float sp = (h - vth >= 0.f) ? 1.f : 0.f;
    s[(long long)t * n + i] = sp;
    v = h * (1.f - sp);
  }
}

__global__ void k_comb1(const float* x, const float* s, const float* hs,
                        const float* ls, float* out, long long n) {
  long long i = (long long)blockIdx.x * blockDim.x + threadIdx.x;
  if (i >= n) return;
  float xv = x[i], sv = s[i];
  out[i] = hs[0] * sv + ls[0] * (xv - sv) + xv * sv;
}

__global__ void k_comb2(const float* o, const float* x2, const float* x1,
                        const float* hs, const float* ls, const float* gate,
                        float* out, long long n) {
  long long i = (long long)blockIdx.x * blockDim.x + threadIdx.x;
  if (i >= n) return;
  float ov = o[i], v2 = x2[i], v1 = x1[i];
  out[i] = hs[0] * v2 + ls[0] * (ov - v2) + ov * v2 + sigf(gate[0]) * v1 * v2;
}

__global__ void k_add(const float* a, const float* b, float* out, long long n) {
  long long i = (long long)blockIdx.x * blockDim.x + threadIdx.x;
  if (i < n) out[i] = a[i] + b[i];
}

// fused pixel_unshuffle(r=2) + 16-step decay/quant4 mem_update recurrence.
// mode 0 -> out [T*B, 4C (c-major,k-minor), Hd, Wd]; mode 1 -> [16, B, C, Hd, Wd]
__global__ void k_psl(const float* __restrict__ x, float* __restrict__ out,
                      int T, int B, int C, int H, int W, int mode) {
  const int Hd = H >> 1, Wd = W >> 1;
  long long n = (long long)B * C * Hd * Wd;
  long long i = (long long)blockIdx.x * blockDim.x + threadIdx.x;
  if (i >= n) return;
  int w2 = (int)(i % Wd);
  int h2 = (int)((i / Wd) % Hd);
  int c  = (int)((i / ((long long)Wd * Hd)) % C);
  int b  = (int)(i / ((long long)Wd * Hd * C));
  float mem = 0.f, spk = 0.f;
  for (int tt = 0; tt < T * 4; ++tt) {
    int t = tt >> 2, k = tt & 3, dy = (k >> 1) & 1, dx = k & 1;
    float xt = x[((((long long)t * B + b) * C + c) * H + 2 * h2 + dy) * W + 2 * w2 + dx];
    mem = (mem - spk) * 0.25f + xt;
    float s = rintf(fminf(fmaxf(mem, 0.f), 4.f)) * 0.25f;
    spk = s;
    long long oidx;
    if (mode == 0)
      oidx = ((((long long)t * B + b) * (C * 4) + c * 4 + k) * Hd + h2) * Wd + w2;
    else
      oidx = ((((long long)tt * B + b) * C + c) * Hd + h2) * Wd + w2;
    out[oidx] = s;
  }
}

// bilinear x2 upsample, half-pixel centers (jax.image.resize 'bilinear')
__global__ void k_up2(const float* __restrict__ in, float* __restrict__ out,
                      int N, int C, int Hd, int Wd) {
  const int H = Hd * 2, W = Wd * 2;
  long long total = (long long)N * C * H * W;
  long long i = (long long)blockIdx.x * blockDim.x + threadIdx.x;
  if (i >= total) return;
  int w = (int)(i % W);
  int h = (int)((i / W) % H);
  long long nc = i / ((long long)H * W);
  float sy = h * 0.5f - 0.25f, sx = w * 0.5f - 0.25f;
  int y0 = (int)floorf(sy), x0 = (int)floorf(sx);
  float fy = sy - y0, fx = sx - x0;
  int y0c = min(max(y0, 0), Hd - 1), y1c = min(max(y0 + 1, 0), Hd - 1);
  int x0c = min(max(x0, 0), Wd - 1), x1c = min(max(x0 + 1, 0), Wd - 1);
  const float* p = in + nc * Hd * Wd;
  float v00 = p[y0c * Wd + x0c], v01 = p[y0c * Wd + x1c];
  float v10 = p[y1c * Wd + x0c], v11 = p[y1c * Wd + x1c];
  out[i] = (1.f - fy) * ((1.f - fx) * v00 + fx * v01) + fy * ((1.f - fx) * v10 + fx * v11);
}

// ---------------------------------------------------------------------------
// tdBN: per-channel mean/var over (T,B,H,W) via atomics, then normalize.
// ---------------------------------------------------------------------------
__device__ __forceinline__ void blockReduceSumMax(float& s, float& m) {
  __shared__ float ss[256], sm[256];
  int tid = threadIdx.x;
  __syncthreads();                       // protect LDS reuse across calls
  ss[tid] = s; sm[tid] = m;
  __syncthreads();
  for (int o = 128; o > 0; o >>= 1) {
    if (tid < o) { ss[tid] += ss[tid + o]; sm[tid] = fmaxf(sm[tid], sm[tid + o]); }
    __syncthreads();
  }
  s = ss[0]; m = sm[0];
}

__global__ void k_bn_stats(const float* __restrict__ x, float* stats,
                           int C, int TB, int HW) {
  const int c = blockIdx.x;
  long long per = (long long)TB * HW;
  float s = 0.f, q = 0.f;
  for (long long i = (long long)blockIdx.y * blockDim.x + threadIdx.x; i < per;
       i += (long long)gridDim.y * blockDim.x) {
    int tb = (int)(i / HW), hw = (int)(i % HW);
    float v = x[((long long)tb * C + c) * HW + hw];
    s += v; q += v * v;
  }
  float dummy = 0.f;
  blockReduceSumMax(s, dummy);
  blockReduceSumMax(q, dummy);
  if (threadIdx.x == 0) { atomicAdd(&stats[c * 2], s); atomicAdd(&stats[c * 2 + 1], q); }
}

__global__ void k_bn_apply(const float* x, float* y, const float* stats,
                           const float* g, const float* b, int C, int HW,
                           long long n, float cnt) {
  long long i = (long long)blockIdx.x * blockDim.x + threadIdx.x;
  if (i >= n) return;
  int c = (int)((i / HW) % C);
  float mean = stats[c * 2] / cnt;
  float var  = stats[c * 2 + 1] / cnt - mean * mean;
  y[i] = (x[i] - mean) * rsqrtf(var + 1e-5f) * g[c] + b[c];
}

// ---------------------------------------------------------------------------
// Attention (MDA / FSTA / TA16) helpers
// ---------------------------------------------------------------------------
__global__ void k_red_contig(const float* __restrict__ x, float* omean,
                             float* omax, long long inner) {
  long long o = blockIdx.x;
  const float* p = x + o * inner;
  float s = 0.f, m = -3.4e38f;
  for (long long i = threadIdx.x; i < inner; i += blockDim.x) {
    float v = p[i]; s += v; m = fmaxf(m, v);
  }
  blockReduceSumMax(s, m);
  if (threadIdx.x == 0) { omean[o] = s / (float)inner; omax[o] = m; }
}

__global__ void k_red_bc(const float* __restrict__ x, float* omean, float* omax,
                         int T, int B, int C, int HW) {
  int bc = blockIdx.x;
  int b = bc / C, c = bc % C;
  float s = 0.f, m = -3.4e38f;
  long long per = (long long)T * HW;
  for (long long i = threadIdx.x; i < per; i += blockDim.x) {
    int t = (int)(i / HW), hw = (int)(i % HW);
    float v = x[(((long long)t * B + b) * C + c) * HW + hw];
    s += v; m = fmaxf(m, v);
  }
  blockReduceSumMax(s, m);
  if (threadIdx.x == 0) { omean[bc] = s / (float)per; omax[bc] = m; }
}

// sigmoid( mlp(mean) + mlp(max) ), shared 2-layer relu MLP; tiny, 1 block.
// v[b][i] = vec[i*sI + b*sB]; out[o*oO + b*oB].
__global__ void k_mlp_sig(const float* meanv, const float* maxv,
                          const float* w1, const float* w2, float* out,
                          int Bn, int In, int Hh, int On,
                          int sI, int sB, int oO, int oB) {
  int idx = blockIdx.x * blockDim.x + threadIdx.x;
  if (idx >= Bn * On) return;
  int b = idx / On, o = idx % On;
  float acc = 0.f;
  for (int pass = 0; pass < 2; ++pass) {
    const float* v = pass ? maxv : meanv;
    if (!v) continue;
    float accp = 0.f;
    for (int h = 0; h < Hh; ++h) {
      float hid = 0.f;
      for (int i = 0; i < In; ++i) hid += w1[h * In + i] * v[i * sI + b * sB];
      hid = fmaxf(hid, 0.f);
      accp += w2[o * Hh + h] * hid;
    }
    acc += accp;
  }
  out[o * oO + b * oB] = sigf(acc);
}

__global__ void k_scale_lead(float* y, const float* s, long long inner, long long n) {
  long long i = (long long)blockIdx.x * blockDim.x + threadIdx.x;
  if (i < n) y[i] *= s[i / inner];
}

__global__ void k_scale_bc(float* y, const float* s, int HW, int C, int B, long long n) {
  long long i = (long long)blockIdx.x * blockDim.x + threadIdx.x;
  if (i >= n) return;
  int c = (int)((i / HW) % C);
  int b = (int)((i / ((long long)HW * C)) % B);
  y[i] *= s[b * C + c];
}

// per-pixel channel mean & max -> s2 [N,2,HW]
__global__ void k_cmm(const float* __restrict__ y, float* __restrict__ s2,
                      int N, int C, int HW) {
  long long total = (long long)N * HW;
  long long i = (long long)blockIdx.x * blockDim.x + threadIdx.x;
  if (i >= total) return;
  int hw = (int)(i % HW);
  long long n = i / HW;
  float s = 0.f, m = -3.4e38f;
  for (int c = 0; c < C; ++c) {
    float v = y[(n * C + c) * HW + hw];
    s += v; m = fmaxf(m, v);
  }
  s2[(n * 2 + 0) * (long long)HW + hw] = s / (float)C;
  s2[(n * 2 + 1) * (long long)HW + hw] = m;
}

// tiny 3x3 conv 2ch->1ch + sigmoid (spatial attention)
__global__ void k_conv21(const float* __restrict__ s2, const float* __restrict__ w,
                         float* __restrict__ sa, int N, int H, int W) {
  long long total = (long long)N * H * W;
  long long i = (long long)blockIdx.x * blockDim.x + threadIdx.x;
  if (i >= total) return;
  int wx = (int)(i % W);
  int hy = (int)((i / W) % H);
  long long n = i / ((long long)H * W);
  float acc = 0.f;
  for (int ci = 0; ci < 2; ++ci)
    for (int kh = 0; kh < 3; ++kh)
      for (int kw = 0; kw < 3; ++kw) {
        int ih = hy + kh - 1, iw = wx + kw - 1;
        if (ih >= 0 && ih < H && iw >= 0 && iw < W)
          acc += w[(ci * 3 + kh) * 3 + kw] * s2[((n * 2 + ci) * H + ih) * (long long)W + iw];
      }
  sa[i] = sigf(acc);
}

// y = relu(y * sa[n,hw]) (+ addx) ; sa per image
__global__ void k_sa_apply(const float* y, const float* sa, const float* addx,
                           float* out, long long n, int HW, int C) {
  long long i = (long long)blockIdx.x * blockDim.x + threadIdx.x;
  if (i >= n) return;
  int hw = (int)(i % HW);
  long long nimg = i / ((long long)HW * C);
  float v = fmaxf(y[i] * sa[nimg * HW + hw], 0.f);
  if (addx) v += addx[i];
  out[i] = v;
}

__global__ void k_mean_T(const float* x, float* f, int T, long long n) {
  long long i = (long long)blockIdx.x * blockDim.x + threadIdx.x;
  if (i >= n) return;
  float s = 0.f;
  for (int t = 0; t < T; ++t) s += x[(long long)t * n + i];
  f[i] = s / (float)T;
}

__global__ void k_fsta_apply(const float* x, const float* ca, const float* sa,
                             float* out, int B, int C, int HW, long long n) {
  long long i = (long long)blockIdx.x * blockDim.x + threadIdx.x;
  if (i >= n) return;
  int hw = (int)(i % HW);
  int c  = (int)((i / HW) % C);
  int b  = (int)((i / ((long long)HW * C)) % B);
  out[i] = x[i] * ca[b * C + c] * sa[(long long)b * HW + hw];
}

// repack down[(t*4+k),b,i,h,w] -> packed[(t*B+b), i*4+k, h, w]
__global__ void k_repack(const float* __restrict__ down, float* __restrict__ pk,
                         int B, int C, int HdWd) {
  long long total = (long long)4 * B * C * 4 * HdWd;
  long long i = (long long)blockIdx.x * blockDim.x + threadIdx.x;
  if (i >= total) return;
  int hw = (int)(i % HdWd);
  int ch = (int)((i / HdWd) % (4 * C));
  int n  = (int)(i / ((long long)HdWd * 4 * C));
  int c = ch >> 2, k = ch & 3;
  int t = n / B, b = n % B;
  pk[i] = down[((((long long)(t * 4 + k) * B + b) * C + c) * (long long)HdWd) + hw];
}

// ---------------------------------------------------------------------------
// Frequency-domain MLP: length-128 radix-2 FFT in LDS (rows + cols), complex
// channel MLP at each retained frequency (j<=64), Hermitian completion, iFFT.
// ---------------------------------------------------------------------------
__device__ __forceinline__ void fft128(float2* s, int tid, float dir) {
  unsigned r = __brev((unsigned)tid) >> 25;   // 7-bit reversal
  if (r > (unsigned)tid) { float2 a = s[tid]; s[tid] = s[r]; s[r] = a; }
  __syncthreads();
  for (int len = 2; len <= 128; len <<= 1) {
    int half = len >> 1;
    if (tid < 64) {
      int grp = tid / half, pos = tid - grp * half;
      int a = grp * len + pos, b = a + half;
      float ang = dir * 6.28318530717958647692f * (float)pos / (float)len;
      float cw = __cosf(ang), sw = __sinf(ang);
      float2 u = s[a], w = s[b];
      float tr = w.x * cw - w.y * sw, ti = w.x * sw + w.y * cw;
      s[a] = make_float2(u.x + tr, u.y + ti);
      s[b] = make_float2(u.x - tr, u.y - ti);
    }
    __syncthreads();
  }
}

__global__ void __launch_bounds__(128) k_fft_rows(const float* __restrict__ x,
                                                  float2* __restrict__ X) {
  __shared__ float2 s[128];
  long long row = blockIdx.x;
  int tid = threadIdx.x;
  s[tid] = make_float2(x[row * 128 + tid], 0.f);
  __syncthreads();
  fft128(s, tid, -1.f);
  X[row * 128 + tid] = s[tid];
}

__global__ void __launch_bounds__(128) k_fft_cols(float2* __restrict__ X,
                                                  int H, int W, float dir) {
  __shared__ float2 s[128];
  long long col = blockIdx.x;
  long long img = col / W;
  int w = (int)(col % W);
  long long base = img * (long long)H * W + w;
  int tid = threadIdx.x;
  s[tid] = X[base + (long long)tid * W];
  __syncthreads();
  fft128(s, tid, dir);
  X[base + (long long)tid * W] = s[tid];
}

__global__ void __launch_bounds__(64)
k_freq_mlp(const float2* __restrict__ X, float2* __restrict__ Y,
           const float* w1r, const float* w1i, const float* b1r, const float* b1i,
           const float* w2r, const float* w2i, const float* b2r, const float* b2i,
           int C, int H, int W, int Wf) {
  __shared__ float xr[32], xi[32], hr[64], hi[64];
  long long p = blockIdx.x;                 // over TB*H*Wf
  int j = (int)(p % Wf);
  long long q = p / Wf;
  int h = (int)(q % H);
  long long tb = q / H;
  int tid = threadIdx.x;
  if (tid < C) {
    float2 v = X[((tb * C + tid) * H + h) * (long long)W + j];
    xr[tid] = v.x; xi[tid] = v.y;
  }
  __syncthreads();
  float ar = b1r[tid], ai = b1i[tid];
  for (int c = 0; c < 32; ++c) {
    float wr = w1r[tid * 32 + c], wi = w1i[tid * 32 + c];
    ar += wr * xr[c] - wi * xi[c];
    ai += wi * xr[c] + wr * xi[c];
  }
  hr[tid] = fmaxf(ar, 0.f); hi[tid] = fmaxf(ai, 0.f);
  __syncthreads();
  if (tid < C) {
    float orr = b2r[tid], oii = b2i[tid];
    for (int h2 = 0; h2 < 64; ++h2) {
      float wr = w2r[tid * 64 + h2], wi = w2i[tid * 64 + h2];
      orr += wr * hr[h2] - wi * hi[h2];
      oii += wi * hr[h2] + wr * hi[h2];
    }
    Y[((tb * C + tid) * H + h) * (long long)W + j] = make_float2(orr, oii);
  }
}

__global__ void k_herm(float2* __restrict__ Y, int M, int H, int W, int Wf) {
  long long total = (long long)M * H * (W - Wf);
  long long i = (long long)blockIdx.x * blockDim.x + threadIdx.x;
  if (i >= total) return;
  int j = Wf + (int)(i % (W - Wf));
  long long q = i / (W - Wf);
  int h = (int)(q % H);
  long long m = q / H;
  float2 v = Y[(m * H + ((H - h) % H)) * (long long)W + (W - j)];
  Y[(m * H + h) * (long long)W + j] = make_float2(v.x, -v.y);
}

__global__ void __launch_bounds__(128)
k_ifft_rows_add(const float2* __restrict__ X, const float* __restrict__ resid,
                float* __restrict__ out) {
  __shared__ float2 s[128];
  long long row = blockIdx.x;
  int tid = threadIdx.x;
  s[tid] = X[row * 128 + tid];
  __syncthreads();
  fft128(s, tid, 1.f);
  out[row * 128 + tid] = resid[row * 128 + tid] + s[tid].x * (1.f / 16384.f);
}

// ---------------------------------------------------------------------------
// Host-side orchestration
// ---------------------------------------------------------------------------
extern "C" void kernel_launch(void* const* d_in, const int* in_sizes, int n_in,
                              void* d_out, int out_size, void* d_ws, size_t ws_size,
                              hipStream_t stream) {
  (void)in_sizes; (void)n_in; (void)out_size; (void)ws_size;
  const int T = 4, B = 4, C = 32, H = 128, W = 128;
  const int HW = H * W, Hd = 64, Wd = 64, HWd = Hd * Wd, TB = T * B;
  const long long FULL = (long long)T * B * C * H * W;   // 8,388,608

  // ---- inputs (setup_inputs dict order, lists flattened in order) ----
  const float* x       = (const float*)d_in[0];
  const float* rb_c1w  = (const float*)d_in[1];
  const float* rb_g1   = (const float*)d_in[2];
  const float* rb_b1   = (const float*)d_in[3];
  const float* rb_hs1  = (const float*)d_in[4];
  const float* rb_ls1  = (const float*)d_in[5];
  const float* rb_adj  = (const float*)d_in[6];
  const float* rb_c2w  = (const float*)d_in[7];
  const float* rb_g2   = (const float*)d_in[8];
  const float* rb_b2   = (const float*)d_in[9];
  const float* rb_hs2  = (const float*)d_in[10];
  const float* rb_ls2  = (const float*)d_in[11];
  const float* rb_gate = (const float*)d_in[12];
  const float* rb_scw  = (const float*)d_in[13];
  const float* rb_scg  = (const float*)d_in[14];
  const float* rb_scb  = (const float*)d_in[15];
  const float* at_taw1 = (const float*)d_in[16];
  const float* at_taw2 = (const float*)d_in[17];
  const float* at_caw1 = (const float*)d_in[18];
  const float* at_caw2 = (const float*)d_in[19];
  const float* at_saw  = (const float*)d_in[20];
  const float* fs_w1   = (const float*)d_in[21];
  const float* fs_w2   = (const float*)d_in[22];
  const float* fs_saw  = (const float*)d_in[23];
  const float* ta16w1  = (const float*)d_in[24];
  const float* ta16w2  = (const float*)d_in[25];
  const float* tc_w    = (const float*)d_in[26];
  const float* c1B_w   = (const float*)d_in[27];
  const float* g1B     = (const float*)d_in[28];
  const float* b1B     = (const float*)d_in[29];
  const float* a1_taw1 = (const float*)d_in[30];
  const float* a1_taw2 = (const float*)d_in[31];
  const float* a1_caw1 = (const float*)d_in[32];
  const float* a1_caw2 = (const float*)d_in[33];
  const float* a1_saw  = (const float*)d_in[34];
  const float* c2B_w   = (const float*)d_in[35];
  const float* g2B     = (const float*)d_in[36];
  const float* b2B     = (const float*)d_in[37];
  const float* a2_taw1 = (const float*)d_in[38];
  const float* a2_taw2 = (const float*)d_in[39];
  const float* a2_caw1 = (const float*)d_in[40];
  const float* a2_caw2 = (const float*)d_in[41];
  const float* a2_saw  = (const float*)d_in[42];
  const float* fa_taw1 = (const float*)d_in[43];
  const float* fa_taw2 = (const float*)d_in[44];
  const float* fa_caw1 = (const float*)d_in[45];
  const float* fa_caw2 = (const float*)d_in[46];
  const float* fa_saw  = (const float*)d_in[47];
  const float* fw1r = (const float*)d_in[48];
  const float* fw1i = (const float*)d_in[49];
  const float* fb1r = (const float*)d_in[50];
  const float* fb1i = (const float*)d_in[51];
  const float* fw2r = (const float*)d_in[52];
  const float* fw2i = (const float*)d_in[53];
  const float* fb2r = (const float*)d_in[54];
  const float* fb2i = (const float*)d_in[55];

  // ---- workspace carve ----
  char* wp = (char*)d_ws;
  auto alloc = [&](size_t nbytes) -> void* {
    void* p = (void*)wp;
    wp += (nbytes + 255) & ~(size_t)255;
    return p;
  };
  float*  XH1  = (float*)alloc(FULL * 4);
  float*  TAb  = (float*)alloc(FULL * 4);      // comb buffers / PACK
  float*  TBb  = (float*)alloc(FULL * 4);      // running "out"
  float*  TCb  = (float*)alloc(FULL * 4);      // xh2 / sc
  float*  RES  = (float*)alloc(FULL * 4);      // residual-block output (skip)
  float*  PS   = (float*)alloc(FULL * 4);      // psl spikes / DOWN
  float*  UP   = (float*)alloc(FULL * 4 * 4);  // 128-ch upsample
  float2* SX   = (float2*)alloc(FULL * 8);
  float2* SY   = (float2*)alloc(FULL * 8);
  float*  Fm   = (float*)alloc((size_t)B * C * HW * 4);
  float*  S2   = (float*)alloc((size_t)TB * 2 * HW * 4);
  float*  SA   = (float*)alloc((size_t)TB * HW * 4);
  float*  Hb1  = (float*)alloc((size_t)TB * C * HWd * 4);
  float*  Hb2  = (float*)alloc((size_t)TB * C * HWd * 4);
  float*  Hb3  = (float*)alloc((size_t)TB * C * HWd * 4);
  float*  STATS = (float*)alloc(64 * 4);
  float*  R1   = (float*)alloc(4096);
  float*  R2   = (float*)alloc(4096);
  float*  V1   = (float*)alloc(4096);
  // packed bf16 weights + activation staging buffer
  unsigned short* Wq_c1  = (unsigned short*)alloc(9216 * 2);
  unsigned short* Wq_c2  = (unsigned short*)alloc(9216 * 2);
  unsigned short* Wq_sc  = (unsigned short*)alloc(9216 * 2);
  unsigned short* Wq_d1  = (unsigned short*)alloc(9216 * 2);
  unsigned short* Wq_d2  = (unsigned short*)alloc(9216 * 2);
  unsigned short* Wq_adj = (unsigned short*)alloc(4096 * 2);
  unsigned short* Wq_tc  = (unsigned short*)alloc(4096 * 2);
  unsigned short* QA     = (unsigned short*)alloc((size_t)TB * HW * 128 * 2); // 67 MB

  auto g1d = [](long long n) { return dim3((unsigned)((n + 255) / 256)); };

  // ---- one-time weight packs (fp32 -> bf16, conv layouts) ----
  k_pack_w3<<<36, 256, 0, stream>>>(rb_c1w, Wq_c1);
  k_pack_w3<<<36, 256, 0, stream>>>(rb_c2w, Wq_c2);
  k_pack_w3<<<36, 256, 0, stream>>>(rb_scw, Wq_sc);
  k_pack_w3<<<36, 256, 0, stream>>>(c1B_w, Wq_d1);
  k_pack_w3<<<36, 256, 0, stream>>>(c2B_w, Wq_d2);
  k_pack_w1<<<16, 256, 0, stream>>>(rb_adj, Wq_adj);
  k_pack_w1<<<16, 256, 0, stream>>>(tc_w, Wq_tc);

  auto run_bn = [&](float* buf, const float* g, const float* bt, int Nimg, int HWy) {
    k_zero<<<1, 64, 0, stream>>>(STATS, 64);
    k_bn_stats<<<dim3(C, 64), 256, 0, stream>>>(buf, STATS, C, Nimg, HWy);
    long long n = (long long)Nimg * C * HWy;
    k_bn_apply<<<g1d(n), 256, 0, stream>>>(buf, buf, STATS, g, bt, C, HWy, n,
                                           (float)((long long)Nimg * HWy));
  };
  auto run_conv3 = [&](const float* in, const unsigned short* wq, float* out,
                       int Hy, int Wy) {
    int HWy = Hy * Wy;
    long long tot = (long long)TB * HWy * 32;
    k_pack_nhwc<<<g1d(tot), 256, 0, stream>>>(in, QA, 32, HWy, tot);
    k_conv3x3_wmma<<<dim3(Wy / 32, Hy, TB), 128, 0, stream>>>(QA, wq, out, Hy, Wy);
  };
  auto run_conv1 = [&](const float* in, const unsigned short* wq, float* out,
                       int Hy, int Wy, int relu) {
    int HWy = Hy * Wy;
    long long tot = (long long)TB * HWy * 128;
    k_pack_nhwc<<<g1d(tot), 256, 0, stream>>>(in, QA, 128, HWy, tot);
    k_conv1x1_wmma<<<dim3(Wy / 32, Hy, TB), 128, 0, stream>>>(QA, wq, out, Hy, Wy, relu);
  };
  auto run_mda = [&](float* y, int Hy, int Wy, const float* taw1, const float* taw2,
                     const float* caw1, const float* caw2, const float* saw,
                     const float* addx) {
    int HWy = Hy * Wy;
    long long nf = (long long)T * B * C * HWy;
    // time attention
    k_red_contig<<<TB, 256, 0, stream>>>(y, R1, R2, (long long)C * HWy);
    k_mlp_sig<<<1, 256, 0, stream>>>(R1, R2, taw1, taw2, V1, B, T, 1, T, B, 1, B, 1);
    k_scale_lead<<<g1d(nf), 256, 0, stream>>>(y, V1, (long long)C * HWy, nf);
    // channel attention
    k_red_bc<<<B * C, 256, 0, stream>>>(y, R1, R2, T, B, C, HWy);
    k_mlp_sig<<<1, 256, 0, stream>>>(R1, R2, caw1, caw2, V1, B, C, 2, C, 1, C, 1, C);
    k_scale_bc<<<g1d(nf), 256, 0, stream>>>(y, V1, HWy, C, B, nf);
    // spatial attention + relu (+ residual add)
    k_cmm<<<g1d((long long)TB * HWy), 256, 0, stream>>>(y, S2, TB, C, HWy);
    k_conv21<<<g1d((long long)TB * HWy), 256, 0, stream>>>(S2, saw, SA, TB, Hy, Wy);
    k_sa_apply<<<g1d(nf), 256, 0, stream>>>(y, SA, addx, y, nf, HWy, C);
  };

  // ================= spiking residual block =================
  long long nsite = (long long)B * C * HW;
  k_lif<<<g1d(nsite), 256, 0, stream>>>(x, XH1, nsite, T, 0.15f);
  k_comb1<<<g1d(FULL), 256, 0, stream>>>(x, XH1, rb_hs1, rb_ls1, TAb, FULL);
  run_conv3(TAb, Wq_c1, TBb, H, W);
  run_bn(TBb, rb_g1, rb_b1, TB, HW);
  // PSL block: unshuffle+mem_update -> bilinear up -> 1x1 adj mix (WMMA)
  k_psl<<<g1d((long long)B * C * HWd), 256, 0, stream>>>(TBb, PS, T, B, C, H, W, 0);
  k_up2<<<g1d((long long)TB * 128 * HW), 256, 0, stream>>>(PS, UP, TB, 4 * C, Hd, Wd);
  run_conv1(UP, Wq_adj, TCb, H, W, 0);
  k_comb2<<<g1d(FULL), 256, 0, stream>>>(TBb, TCb, XH1, rb_hs2, rb_ls2, rb_gate, TAb, FULL);
  run_conv3(TAb, Wq_c2, TBb, H, W);
  run_bn(TBb, rb_g2, rb_b2, TB, HW);
  run_conv3(x, Wq_sc, TCb, H, W);
  run_bn(TCb, rb_scg, rb_scb, TB, HW);
  k_add<<<g1d(FULL), 256, 0, stream>>>(TBb, TCb, TBb, FULL);
  run_mda(TBb, H, W, at_taw1, at_taw2, at_caw1, at_caw2, at_saw, x);  // + x
  // FSTA
  k_mean_T<<<g1d(nsite), 256, 0, stream>>>(TBb, Fm, T, nsite);
  k_red_contig<<<B * C, 256, 0, stream>>>(Fm, R1, R2, (long long)HW);
  k_mlp_sig<<<1, 256, 0, stream>>>(R1, nullptr, fs_w1, fs_w2, V1, B, C, 2, C, 1, C, 1, C);
  k_cmm<<<g1d((long long)B * HW), 256, 0, stream>>>(Fm, S2, B, C, HW);
  k_conv21<<<g1d((long long)B * HW), 256, 0, stream>>>(S2, fs_saw, SA, B, H, W);
  k_fsta_apply<<<g1d(FULL), 256, 0, stream>>>(TBb, V1, SA, RES, B, C, HW, FULL);

  // ================= down path =================
  k_psl<<<g1d((long long)B * C * HWd), 256, 0, stream>>>(RES, PS, T, B, C, H, W, 1);
  // 16-step temporal attention
  k_red_contig<<<16 * B, 256, 0, stream>>>(PS, R1, R2, (long long)C * HWd);
  k_mlp_sig<<<1, 256, 0, stream>>>(R1, R2, ta16w1, ta16w2, V1, B, 16, 4, 16, B, 1, B, 1);
  k_scale_lead<<<g1d(FULL), 256, 0, stream>>>(PS, V1, (long long)C * HWd, FULL);
  // temporal compress conv3d -> 1x1 GEMM with K=128 (WMMA, relu)
  k_repack<<<g1d(FULL), 256, 0, stream>>>(PS, TAb, B, C, HWd);
  run_conv1(TAb, Wq_tc, Hb1, Hd, Wd, 1);
  // lif -> conv -> bn -> mda, twice, at 64x64
  long long nsd = (long long)B * C * HWd;
  k_lif<<<g1d(nsd), 256, 0, stream>>>(Hb1, Hb2, nsd, 4, 0.15f);
  run_conv3(Hb2, Wq_d1, Hb3, Hd, Wd);
  run_bn(Hb3, g1B, b1B, TB, HWd);
  run_mda(Hb3, Hd, Wd, a1_taw1, a1_taw2, a1_caw1, a1_caw2, a1_saw, nullptr);
  k_lif<<<g1d(nsd), 256, 0, stream>>>(Hb3, Hb2, nsd, 4, 0.15f);
  run_conv3(Hb2, Wq_d2, Hb1, Hd, Wd);
  run_bn(Hb1, g2B, b2B, TB, HWd);
  run_mda(Hb1, Hd, Wd, a2_taw1, a2_taw2, a2_caw1, a2_caw2, a2_saw, nullptr);

  // ================= up + fuse + final attention =================
  k_up2<<<g1d(FULL), 256, 0, stream>>>(Hb1, TBb, TB, C, Hd, Wd);
  k_add<<<g1d(FULL), 256, 0, stream>>>(TBb, RES, TBb, FULL);
  run_mda(TBb, H, W, fa_taw1, fa_taw2, fa_caw1, fa_caw2, fa_saw, nullptr);

  // ================= frequency-domain MLP (rfft2 / c-MLP / irfft2) =========
  const int M = TB * C;        // 512 planes
  const int Wf = W / 2 + 1;    // 65 retained frequencies
  k_fft_rows<<<(unsigned)((long long)M * H), 128, 0, stream>>>(TBb, SX);
  k_fft_cols<<<(unsigned)((long long)M * W), 128, 0, stream>>>(SX, H, W, -1.f);
  k_freq_mlp<<<(unsigned)((long long)TB * H * Wf), 64, 0, stream>>>(
      SX, SY, fw1r, fw1i, fb1r, fb1i, fw2r, fw2i, fb2r, fb2i, C, H, W, Wf);
  k_herm<<<g1d((long long)M * H * (W - Wf)), 256, 0, stream>>>(SY, M, H, W, Wf);
  k_fft_cols<<<(unsigned)((long long)M * W), 128, 0, stream>>>(SY, H, W, 1.f);
  k_ifft_rows_add<<<(unsigned)((long long)M * H), 128, 0, stream>>>(SY, TBb, (float*)d_out);
}
